// SwinTransformerBlock2D_37417755083133
// MI455X (gfx1250) — matrix operand order repaired and checked by reading, not compiled, not run on hardware
//
#include <hip/hip_runtime.h>
#include <hip/hip_bf16.h>
#include <math.h>

// ---------------------------------------------------------------------------
// Types for CDNA5 WMMA
// ---------------------------------------------------------------------------
typedef __attribute__((ext_vector_type(16))) _Float16 v16h;
typedef __attribute__((ext_vector_type(8)))  _Float16 h8;
typedef __attribute__((ext_vector_type(8)))  float    v8f;
typedef __attribute__((ext_vector_type(4)))  int      v4i;

union U16 { v16h v; h8 h[2]; };

#define TOK   65536          // B * H * W
#define DIMC  512
#define HEADS 16
#define HD    32
#define WS    8
#define SHIFT 4
#define NWIN  1024           // B * (64/8)^2
#define HIDDEN 2048

// ---------------------------------------------------------------------------
// Async global->LDS staging (CDNA5 GLOBAL_LOAD_ASYNC_TO_LDS_B128, ASYNCcnt),
// with a synchronous fallback if the toolchain lacks the builtins.
// Probe result: builtin takes (int4 addrspace(1)*, int4 addrspace(3)*, imm, imm).
// ---------------------------------------------------------------------------
#if defined(__gfx1250__) && \
    __has_builtin(__builtin_amdgcn_global_load_async_to_lds_b128) && \
    __has_builtin(__builtin_amdgcn_s_wait_asynccnt)
#define USE_ASYNC_LDS 1
#else
#define USE_ASYNC_LDS 0
#endif

__device__ __forceinline__ void stage_b128(const _Float16* gsrc, _Float16* ldst) {
#if USE_ASYNC_LDS
  __builtin_amdgcn_global_load_async_to_lds_b128(
      (__attribute__((address_space(1))) v4i*)gsrc,
      (__attribute__((address_space(3))) v4i*)ldst, 0, 0);
#else
  *(h8*)ldst = *(const h8*)gsrc;
#endif
}

__device__ __forceinline__ void stage_wait() {
#if USE_ASYNC_LDS
  __builtin_amdgcn_s_wait_asynccnt(0);
#endif
}

// ---------------------------------------------------------------------------
// f32 -> f16 weight conversion
// ---------------------------------------------------------------------------
__global__ __launch_bounds__(256) void cvt_kernel(const float* __restrict__ src,
                                                  _Float16* __restrict__ dst, int n) {
  int i = blockIdx.x * 256 + threadIdx.x;
  if (i < n) dst[i] = (_Float16)src[i];
}

// ---------------------------------------------------------------------------
// LayerNorm (one block of 256 threads per 512-wide row), f32 in -> f16 out
// ---------------------------------------------------------------------------
__global__ __launch_bounds__(256) void ln_kernel(const float* __restrict__ x,
                                                 const float* __restrict__ g,
                                                 const float* __restrict__ b,
                                                 _Float16* __restrict__ y) {
  __shared__ float red[2][8];
  const int row = blockIdx.x;
  const int tid = threadIdx.x;
  const float* xr = x + (size_t)row * DIMC;
  float v0 = xr[tid], v1 = xr[tid + 256];
  float s = v0 + v1, q = v0 * v0 + v1 * v1;
#pragma unroll
  for (int off = 16; off >= 1; off >>= 1) {
    s += __shfl_xor(s, off, 32);
    q += __shfl_xor(q, off, 32);
  }
  const int wave = tid >> 5, lane = tid & 31;
  if (lane == 0) { red[0][wave] = s; red[1][wave] = q; }
  __syncthreads();
  float ts = 0.f, tq = 0.f;
#pragma unroll
  for (int i = 0; i < 8; ++i) { ts += red[0][i]; tq += red[1][i]; }
  const float mean = ts * (1.0f / DIMC);
  const float var  = tq * (1.0f / DIMC) - mean * mean;
  const float rstd = rsqrtf(var + 1e-5f);
  _Float16* yr = y + (size_t)row * DIMC;
  yr[tid]       = (_Float16)((v0 - mean) * rstd * g[tid]       + b[tid]);
  yr[tid + 256] = (_Float16)((v1 - mean) * rstd * g[tid + 256] + b[tid + 256]);
}

// ---------------------------------------------------------------------------
// Tiled WMMA GEMM: C[M,N] = act[M,K] @ W[N,K]^T + bias, with epilogues
//   EPI 0: store f16
//   EPI 1: store f16 exact-GELU
//   EPI 2: store f32 residual + value
// Block tile 128x64, 8 waves, each wave 16x64, K-step 32.
// Double-buffered LDS with async global->LDS staging (ASYNCcnt pipeline).
// ---------------------------------------------------------------------------
template <int EPI>
__global__ __launch_bounds__(256)
void gemm_wmma_kernel(const _Float16* __restrict__ A, const _Float16* __restrict__ Bw,
                      const float* __restrict__ bias, int M, int N, int K,
                      float* __restrict__ outf, _Float16* __restrict__ outh,
                      const float* __restrict__ resid) {
  __shared__ _Float16 As[2][128 * 32];
  __shared__ _Float16 Bs[2][64 * 32];
  const int tid  = threadIdx.x;
  const int wave = tid >> 5;
  const int lane = tid & 31;
  const int lq   = lane & 15;
  const int kh   = lane >> 4;
  const int m0   = blockIdx.y * 128;
  const int n0   = blockIdx.x * 64;

  v8f acc[4] = {};

  const int ldr = tid >> 2;            // 0..63
  const int ldc = (tid & 3) * 8;       // 0,8,16,24

  // Issue one block tile (3 x b128 per thread) into buffer `buf`.
  auto stage_tile = [&](int buf, int k0) {
    stage_b128(&A[(size_t)(m0 + ldr)      * K + k0 + ldc], &As[buf][(ldr)      * 32 + ldc]);
    stage_b128(&A[(size_t)(m0 + ldr + 64) * K + k0 + ldc], &As[buf][(ldr + 64) * 32 + ldc]);
    stage_b128(&Bw[(size_t)(n0 + ldr)     * K + k0 + ldc], &Bs[buf][(ldr)      * 32 + ldc]);
  };

  stage_tile(0, 0);

  int buf = 0;
  for (int k0 = 0; k0 < K; k0 += 32, buf ^= 1) {
    stage_wait();        // own async loads into `buf` complete
    __syncthreads();     // all waves' loads complete; previous compute done
    if (k0 + 32 < K) {
      stage_tile(buf ^ 1, k0 + 32);
      if (k0 + 64 < K)   // warm GL2 two steps ahead
        __builtin_prefetch(&Bw[(size_t)(n0 + ldr) * K + k0 + 64 + ldc], 0, 0);
    }

    // A fragment (16x32): lane holds row m, K split per ISA layout.
    const int m = wave * 16 + lq;
    U16 a;
    a.h[0] = *(const h8*)&As[buf][m * 32 + kh * 8];
    a.h[1] = *(const h8*)&As[buf][m * 32 + 16 + kh * 8];

#pragma unroll
    for (int nt = 0; nt < 4; ++nt) {
      const int n = nt * 16 + lq;
      U16 bf;
      bf.h[0] = *(const h8*)&Bs[buf][n * 32 + kh * 8];
      bf.h[1] = *(const h8*)&Bs[buf][n * 32 + 16 + kh * 8];
      acc[nt] = __builtin_amdgcn_wmma_f32_16x16x32_f16(
          false, a.v, false, bf.v, (short)0, acc[nt], false, false);
    }
  }

  // Epilogue: C layout -> lane lq = col, VGPR r -> row r + 8*kh.
#pragma unroll
  for (int nt = 0; nt < 4; ++nt) {
    const int n = n0 + nt * 16 + lq;
    const float bv = bias[n];
#pragma unroll
    for (int r = 0; r < 8; ++r) {
      const int m = m0 + wave * 16 + r + kh * 8;
      const float v = acc[nt][r] + bv;
      const size_t off = (size_t)m * N + n;
      if (EPI == 0) {
        outh[off] = (_Float16)v;
      } else if (EPI == 1) {
        outh[off] = (_Float16)(0.5f * v * (1.0f + erff(v * 0.70710678118f)));
      } else {
        outf[off] = resid[off] + v;
      }
    }
  }
}

// ---------------------------------------------------------------------------
// Windowed attention: one wave per (window, head).
// S = (Q*scale) K^T + rpb + 0.5*(Mi+Mj) + shift_mask ; softmax ; O = P V
// Shift/roll folded into token index math; output written back token-major.
// ---------------------------------------------------------------------------
__global__ __launch_bounds__(32)
void attn_kernel(const _Float16* __restrict__ qkv, const float* __restrict__ M_map,
                 const float* __restrict__ rpb_table, _Float16* __restrict__ attn_out) {
  __shared__ _Float16 Plds[64 * 64];   // softmaxed probabilities (f16)
  __shared__ _Float16 Vt[32 * 64];     // V transposed: [hd][token]
  __shared__ float mwin[64];
  __shared__ int   regid[64];
  __shared__ int   tokid[64];

  const int wid  = blockIdx.x;        // 0..1023
  const int head = blockIdx.y;        // 0..15
  const int lane = threadIdx.x;
  const int lq   = lane & 15;
  const int kh   = lane >> 4;
  const int b    = wid >> 6;
  const int wl   = wid & 63;
  const int wh   = wl >> 3, ww = wl & 7;

  // Token mapping, prior-mask values, shift-mask region ids.
  for (int i = lane; i < 64; i += 32) {
    const int lr = i >> 3, lc = i & 7;
    const int hp = wh * WS + lr;                 // rolled coords
    const int wp = ww * WS + lc;
    const int hs = (hp + SHIFT) & 63;            // source (unrolled) coords
    const int wsrc = (wp + SHIFT) & 63;
    const int tok = b * 4096 + hs * 64 + wsrc;
    tokid[i] = tok;
    mwin[i]  = fminf(fmaxf(M_map[tok], 0.0f), 1.0f);
    const int rr = (hp < 56) ? 0 : (hp < 60 ? 1 : 2);
    const int rc = (wp < 56) ? 0 : (wp < 60 ? 1 : 2);
    regid[i] = rr * 3 + rc;
  }
  __syncthreads();

  // Stage V transposed into LDS: Vt[c][k] = V[token k][hd c].
  for (int i = lane; i < 64; i += 32) {
    const _Float16* vr = qkv + (size_t)tokid[i] * 1536 + 1024 + head * HD;
#pragma unroll
    for (int c = 0; c < HD; ++c) Vt[c * 64 + i] = vr[c];
  }

  // ---- S = Q K^T (M=64, N=64, K=32): fragments straight from global ----
  v16h afrag[4], bfrag[4];
#pragma unroll
  for (int mt = 0; mt < 4; ++mt) {
    const int m = mt * 16 + lq;
    const _Float16* qr = qkv + (size_t)tokid[m] * 1536 + head * HD;
    U16 u;
    u.h[0] = *(const h8*)(qr + kh * 8);
    u.h[1] = *(const h8*)(qr + 16 + kh * 8);
    afrag[mt] = u.v;
  }
#pragma unroll
  for (int nt = 0; nt < 4; ++nt) {
    const int n = nt * 16 + lq;
    const _Float16* kr = qkv + (size_t)tokid[n] * 1536 + 512 + head * HD;
    U16 u;
    u.h[0] = *(const h8*)(kr + kh * 8);
    u.h[1] = *(const h8*)(kr + 16 + kh * 8);
    bfrag[nt] = u.v;
  }

  v8f acc[4][4] = {};
#pragma unroll
  for (int mt = 0; mt < 4; ++mt)
#pragma unroll
    for (int nt = 0; nt < 4; ++nt)
      acc[mt][nt] = __builtin_amdgcn_wmma_f32_16x16x32_f16(
          false, afrag[mt], false, bfrag[nt], (short)0, acc[mt][nt], false, false);

  // ---- biases + softmax (row lives in one half-wave under C layout) ----
  const float scale = 0.17677669529f;  // 1/sqrt(32)
#pragma unroll
  for (int mt = 0; mt < 4; ++mt) {
#pragma unroll
    for (int r = 0; r < 8; ++r) {
      const int m  = mt * 16 + r + kh * 8;
      const int ri = m >> 3, ci = m & 7;
      const float mi = mwin[m];
      const int rgi  = regid[m];
      float vals[4];
#pragma unroll
      for (int nt = 0; nt < 4; ++nt) {
        const int n  = nt * 16 + lq;
        const int rj = n >> 3, cj = n & 7;
        const int idx = (ri - rj + 7) * 15 + (ci - cj + 7);
        float bias = rpb_table[idx * HEADS + head] + 0.5f * (mi + mwin[n]);
        if (rgi != regid[n]) bias -= 100.0f;
        vals[nt] = acc[mt][nt][r] * scale + bias;
      }
      float mx = fmaxf(fmaxf(vals[0], vals[1]), fmaxf(vals[2], vals[3]));
#pragma unroll
      for (int off = 8; off >= 1; off >>= 1) mx = fmaxf(mx, __shfl_xor(mx, off, 16));
      float sum = 0.f;
#pragma unroll
      for (int nt = 0; nt < 4; ++nt) { vals[nt] = __expf(vals[nt] - mx); sum += vals[nt]; }
#pragma unroll
      for (int off = 8; off >= 1; off >>= 1) sum += __shfl_xor(sum, off, 16);
      const float inv = 1.0f / sum;
#pragma unroll
      for (int nt = 0; nt < 4; ++nt)
        Plds[m * 64 + nt * 16 + lq] = (_Float16)(vals[nt] * inv);
    }
  }
  __syncthreads();

  // ---- O = P V (M=64, N=32, K=64): A from Plds, B from Vt (contiguous) ----
  v8f o[4][2] = {};
#pragma unroll
  for (int ks = 0; ks < 2; ++ks) {
    v16h pa[4], vb[2];
#pragma unroll
    for (int mt = 0; mt < 4; ++mt) {
      const _Float16* pr = &Plds[(mt * 16 + lq) * 64 + ks * 32];
      U16 u;
      u.h[0] = *(const h8*)(pr + kh * 8);
      u.h[1] = *(const h8*)(pr + 16 + kh * 8);
      pa[mt] = u.v;
    }
#pragma unroll
    for (int nt = 0; nt < 2; ++nt) {
      const _Float16* vr = &Vt[(nt * 16 + lq) * 64 + ks * 32];
      U16 u;
      u.h[0] = *(const h8*)(vr + kh * 8);
      u.h[1] = *(const h8*)(vr + 16 + kh * 8);
      vb[nt] = u.v;
    }
#pragma unroll
    for (int mt = 0; mt < 4; ++mt)
#pragma unroll
      for (int nt = 0; nt < 2; ++nt)
        o[mt][nt] = __builtin_amdgcn_wmma_f32_16x16x32_f16(
            false, pa[mt], false, vb[nt], (short)0, o[mt][nt], false, false);
  }

  // ---- write back (window-reverse + roll is the identical token map) ----
#pragma unroll
  for (int mt = 0; mt < 4; ++mt) {
#pragma unroll
    for (int r = 0; r < 8; ++r) {
      const int m = mt * 16 + r + kh * 8;
      _Float16* orow = attn_out + (size_t)tokid[m] * DIMC + head * HD;
#pragma unroll
      for (int nt = 0; nt < 2; ++nt)
        orow[nt * 16 + lq] = (_Float16)o[mt][nt][r];
    }
  }
}

// ---------------------------------------------------------------------------
// Host-side launch
// ---------------------------------------------------------------------------
extern "C" void kernel_launch(void* const* d_in, const int* in_sizes, int n_in,
                              void* d_out, int out_size, void* d_ws, size_t ws_size,
                              hipStream_t stream) {
  const float* x       = (const float*)d_in[0];
  const float* M_map   = (const float*)d_in[1];
  const float* norm1_g = (const float*)d_in[2];
  const float* norm1_b = (const float*)d_in[3];
  const float* qkv_w   = (const float*)d_in[4];
  const float* qkv_b   = (const float*)d_in[5];
  const float* proj_w  = (const float*)d_in[6];
  const float* proj_b  = (const float*)d_in[7];
  const float* rpb     = (const float*)d_in[8];
  const float* norm2_g = (const float*)d_in[9];
  const float* norm2_b = (const float*)d_in[10];
  const float* fc1_w   = (const float*)d_in[11];
  const float* fc1_b   = (const float*)d_in[12];
  const float* fc2_w   = (const float*)d_in[13];
  const float* fc2_b   = (const float*)d_in[14];
  float* out = (float*)d_out;

  // Workspace carve-up (256-byte aligned slices).
  char* ws = (char*)d_ws;
  size_t o = 0;
  auto carve = [&](size_t bytes) { size_t r = o; o += (bytes + 255) & ~(size_t)255; return r; };
  _Float16* qkvw16 = (_Float16*)(ws + carve((size_t)1536 * 512 * 2));
  _Float16* projw16= (_Float16*)(ws + carve((size_t)512 * 512 * 2));
  _Float16* fc1w16 = (_Float16*)(ws + carve((size_t)HIDDEN * 512 * 2));
  _Float16* fc2w16 = (_Float16*)(ws + carve((size_t)512 * HIDDEN * 2));
  _Float16* act16  = (_Float16*)(ws + carve((size_t)TOK * DIMC * 2));   // LN1/LN2 output
  _Float16* big    = (_Float16*)(ws + carve((size_t)TOK * HIDDEN * 2)); // qkv aliases hidden
  _Float16* attn16 = (_Float16*)(ws + carve((size_t)TOK * DIMC * 2));
  float*    x1     = (float*)   (ws + carve((size_t)TOK * DIMC * 4));
  _Float16* qkvbuf = big;   // TOK x 1536 (fits inside TOK x 2048)
  _Float16* hbuf   = big;   // TOK x 2048 (reused after attention)

  // 1) weight conversion
  cvt_kernel<<<(1536 * 512 + 255) / 256, 256, 0, stream>>>(qkv_w, qkvw16, 1536 * 512);
  cvt_kernel<<<(512 * 512 + 255) / 256, 256, 0, stream>>>(proj_w, projw16, 512 * 512);
  cvt_kernel<<<(HIDDEN * 512 + 255) / 256, 256, 0, stream>>>(fc1_w, fc1w16, HIDDEN * 512);
  cvt_kernel<<<(512 * HIDDEN + 255) / 256, 256, 0, stream>>>(fc2_w, fc2w16, 512 * HIDDEN);

  // 2) LN1
  ln_kernel<<<TOK, 256, 0, stream>>>(x, norm1_g, norm1_b, act16);

  // 3) QKV GEMM: (TOK,512) x (1536,512)^T -> f16
  gemm_wmma_kernel<0><<<dim3(1536 / 64, TOK / 128), 256, 0, stream>>>(
      act16, qkvw16, qkv_b, TOK, 1536, 512, nullptr, qkvbuf, nullptr);

  // 4) windowed attention
  attn_kernel<<<dim3(NWIN, HEADS), 32, 0, stream>>>(qkvbuf, M_map, rpb, attn16);

  // 5) proj GEMM + residual -> x1 (f32)
  gemm_wmma_kernel<2><<<dim3(512 / 64, TOK / 128), 256, 0, stream>>>(
      attn16, projw16, proj_b, TOK, 512, 512, x1, nullptr, x);

  // 6) LN2
  ln_kernel<<<TOK, 256, 0, stream>>>(x1, norm2_g, norm2_b, act16);

  // 7) fc1 GEMM + GELU -> f16 hidden
  gemm_wmma_kernel<1><<<dim3(HIDDEN / 64, TOK / 128), 256, 0, stream>>>(
      act16, fc1w16, fc1_b, TOK, HIDDEN, 512, nullptr, hbuf, nullptr);

  // 8) fc2 GEMM + residual -> d_out (f32)
  gemm_wmma_kernel<2><<<dim3(512 / 64, TOK / 128), 256, 0, stream>>>(
      hbuf, fc2w16, fc2_b, TOK, 512, HIDDEN, out, nullptr, x1);
}
